// BoundaryAttention_10771777978975
// MI455X (gfx1250) — matrix operand order repaired and checked by reading, not compiled, not run on hardware
//
#include <hip/hip_runtime.h>
#include <math.h>

typedef __attribute__((ext_vector_type(16))) __bf16 v16bf;
typedef __attribute__((ext_vector_type(8)))  __bf16 v8bf;
typedef __attribute__((ext_vector_type(8)))  float  v8f;

static constexpr int Bn  = 32;
static constexpr int Cc  = 256;
static constexpr int Nn  = 16384;   // H*W = 128*128
static constexpr int HDn = 64;
static constexpr int NHn = 4;

__device__ __forceinline__ unsigned int pk_bf16(float a, float b) {
  union { __bf16 h; unsigned short u; } ua, ub;
  ua.h = (__bf16)a; ub.h = (__bf16)b;
  return (unsigned int)ua.u | ((unsigned int)ub.u << 16);
}

// ---------------------------------------------------------------------------
// K1: tiny projections  q -> qh -> kq[b,h,64]; bf16 conversions of conv_w, w1
// ---------------------------------------------------------------------------
__global__ __launch_bounds__(256) void k1_prep(
    const float* __restrict__ idemb, const float* __restrict__ idp_w,
    const float* __restrict__ idp_b, const float* __restrict__ wq,
    const float* __restrict__ bq,    const float* __restrict__ wk,
    const float* __restrict__ conv_w, const float* __restrict__ w1,
    float* __restrict__ kq_ws, __bf16* __restrict__ convw_bf,
    __bf16* __restrict__ w1_bf)
{
  __shared__ float s_q[Bn * HDn];
  __shared__ float s_qh[Bn * HDn];
  const int t = threadIdx.x;

  for (int o = t; o < Bn * HDn; o += 256) {           // q = id @ idp_w.T + idp_b
    int b = o >> 6, d = o & 63;
    float s = idp_b[d];
    for (int c = 0; c < 256; ++c) s += idemb[b * 256 + c] * idp_w[d * 256 + c];
    s_q[o] = s;
  }
  __syncthreads();
  for (int o = t; o < Bn * HDn; o += 256) {           // qh = q @ wq.T + bq
    int b = o >> 6, d = o & 63;
    float s = bq[d];
    for (int e = 0; e < 64; ++e) s += s_q[b * 64 + e] * wq[d * 64 + e];
    s_qh[o] = s;
  }
  __syncthreads();
  for (int o = t; o < Bn * NHn * HDn; o += 256) {     // kq[b,h,e] = sum_j qh[b,h,j] wk[h*16+j, e]
    int b = o >> 8, h = (o >> 6) & 3, e = o & 63;
    float s = 0.f;
    for (int j = 0; j < 16; ++j) s += s_qh[b * 64 + h * 16 + j] * wk[(h * 16 + j) * 64 + e];
    kq_ws[o] = s;
  }
  for (int i = t; i < HDn * Cc; i += 256) convw_bf[i] = (__bf16)conv_w[i];
  for (int i = t; i < HDn * HDn; i += 256) w1_bf[i] = (__bf16)w1[i];
}

// ---------------------------------------------------------------------------
// K2: pf = conv1x1(X) via bf16 WMMA (K=256), fused masked score computation.
// pf stored as bf16 (LayerNorm + f32-accumulated attention absorb rounding).
// grid (N/128, B), block 256 (8 waves, one 16-row m-strip each)
// ---------------------------------------------------------------------------
__global__ __launch_bounds__(256) void k2_pf_scores(
    const float* __restrict__ X, const __bf16* __restrict__ Wb,
    const float* __restrict__ conv_b, const float* __restrict__ kq,
    const int* __restrict__ cmask,
    __bf16* __restrict__ pf, float* __restrict__ scores)
{
  __shared__ __align__(16) __bf16 sx[128 * 72];   // [n][c] tile, 144B rows (16B aligned)
  const int t = threadIdx.x;
  const int b = blockIdx.y;
  const int n0 = blockIdx.x * 128;
  const int w = t >> 5, lane = t & 31, lr = lane & 15, hh = lane >> 4;
  const int row = w * 16 + lr;

  v8f acc[4] = {};
  for (int kc = 0; kc < 4; ++kc) {
    // stage X[b, kc*64 .. +63, n0 .. n0+127] -> sx[n][c] bf16, transposed.
    // Each thread: 4 consecutive c for one n -> 2 packed dwords -> ds_store_b64
    for (int i = 0; i < 8; ++i) {
      int idx = t + i * 256;               // 0..2047 quads
      int nl = idx & 127, c4 = (idx >> 7) * 4;
      const float* xp = X + ((size_t)(b * Cc + kc * 64 + c4)) * Nn + n0 + nl;
      float x0 = xp[0];
      float x1 = xp[Nn];
      float x2 = xp[2 * Nn];
      float x3 = xp[3 * Nn];
      uint2 p;
      p.x = pk_bf16(x0, x1);
      p.y = pk_bf16(x2, x3);
      *(uint2*)(&sx[nl * 72 + c4]) = p;
    }
    __syncthreads();
#pragma unroll
    for (int kk = 0; kk < 2; ++kk) {
      // A fragment: 16x32 bf16, lanes 0-15: K 0..15; lanes 16-31: K 16..31
      int abase = row * 72 + kk * 32 + hh * 16;
      union { struct { v8bf lo, hi; } p; v16bf v; } ua;
      ua.p.lo = *(const v8bf*)(&sx[abase]);
      ua.p.hi = *(const v8bf*)(&sx[abase + 8]);
#pragma unroll
      for (int dt = 0; dt < 4; ++dt) {
        // B fragment: lane holds output column d = dt*16+lr, 16 contiguous K
        const v16bf bf = *(const v16bf*)(&Wb[(dt * 16 + lr) * 256 + kc * 64 + kk * 32 + hh * 16]);
        acc[dt] = __builtin_amdgcn_wmma_f32_16x16x32_bf16(
            false, ua.v, false, bf, (short)0, acc[dt], false, false);
      }
    }
    __syncthreads();
  }

  // add conv bias
#pragma unroll
  for (int dt = 0; dt < 4; ++dt) {
    float bias = conv_b[dt * 16 + lr];
#pragma unroll
    for (int r = 0; r < 8; ++r) acc[dt][r] += bias;
  }

  // store pf as bf16: pair adjacent d-columns across lanes, even lanes store b32
#pragma unroll
  for (int dt = 0; dt < 4; ++dt)
#pragma unroll
    for (int r = 0; r < 8; ++r) {
      float vself = acc[dt][r];
      float vnb = __shfl_xor(vself, 1, 32);      // neighbor column d^1
      if ((lr & 1) == 0) {
        int n = n0 + w * 16 + r + hh * 8;
        unsigned int pk = pk_bf16(vself, vnb);
        *(unsigned int*)(&pf[((size_t)b * Nn + n) * 64 + dt * 16 + lr]) = pk;
      }
    }

  // fused scores: score[b,h,n] = scale * pf[n,:].kq[b,h,:]  (masked)
  float kqv[4][4];
  for (int h = 0; h < 4; ++h)
    for (int dt = 0; dt < 4; ++dt)
      kqv[h][dt] = kq[(b * 4 + h) * 64 + dt * 16 + lr];
  const float scale = 0.25f;  // 1/sqrt(16)
  for (int h = 0; h < 4; ++h) {
#pragma unroll
    for (int r = 0; r < 8; ++r) {
      float s = 0.f;
#pragma unroll
      for (int dt = 0; dt < 4; ++dt) s += acc[dt][r] * kqv[h][dt];
      s += __shfl_xor(s, 1, 32);
      s += __shfl_xor(s, 2, 32);
      s += __shfl_xor(s, 4, 32);
      s += __shfl_xor(s, 8, 32);   // reduce over 16 N-lanes (halves stay separate)
      if (lr == 0) {
        int n = n0 + w * 16 + r + hh * 8;
        scores[((size_t)b * 4 + h) * Nn + n] = cmask[n] ? s * scale : -1e9f;
      }
    }
  }
}

// ---------------------------------------------------------------------------
// K3a: per-(b,h) softmax stats (max, sumexp); also zero wsum for this (b,h)
// ---------------------------------------------------------------------------
__global__ __launch_bounds__(256) void k3a_stats(
    const float* __restrict__ scores, float* __restrict__ stats,
    float* __restrict__ wsum)
{
  __shared__ float red[256];
  const int bh = blockIdx.x, t = threadIdx.x;
  const float* s = scores + (size_t)bh * Nn;
  float m = -3.0e38f;
  for (int i = t; i < Nn; i += 256) m = fmaxf(m, s[i]);
  red[t] = m; __syncthreads();
  for (int o = 128; o > 0; o >>= 1) { if (t < o) red[t] = fmaxf(red[t], red[t + o]); __syncthreads(); }
  const float mx = red[0]; __syncthreads();
  float p = 0.f;
  for (int i = t; i < Nn; i += 256) p += expf(s[i] - mx);
  red[t] = p; __syncthreads();
  for (int o = 128; o > 0; o >>= 1) { if (t < o) red[t] += red[t + o]; __syncthreads(); }
  if (t == 0) { stats[bh * 2] = mx; stats[bh * 2 + 1] = red[0]; }
  if (t < 64) wsum[bh * 64 + t] = 0.f;
}

// ---------------------------------------------------------------------------
// K3b: wsum[b,h,:] = sum_n attn[b,h,n] * pf[b,n,:]  (pf read once total, bf16)
// grid (8 n-chunks, B), block 256: thread = (d = t&63, group g = t>>6)
// ---------------------------------------------------------------------------
__global__ __launch_bounds__(256) void k3b_wsum(
    const float* __restrict__ scores, const __bf16* __restrict__ pf,
    const float* __restrict__ stats, float* __restrict__ wsum)
{
  __shared__ float sw[256 * 4];
  const int b = blockIdx.y, chunk = blockIdx.x, t = threadIdx.x;
  const int d = t & 63, g = t >> 6;
  float mx[4], inv[4];
  for (int h = 0; h < 4; ++h) {
    mx[h] = stats[(b * 4 + h) * 2];
    inv[h] = 1.f / stats[(b * 4 + h) * 2 + 1];
  }
  float acc[4] = {0.f, 0.f, 0.f, 0.f};
  for (int it = 0; it < 8; ++it) {
    const int nb = chunk * 2048 + it * 256;
    {
      int n = nb + t;
      for (int h = 0; h < 4; ++h)
        sw[t * 4 + h] = expf(scores[((size_t)b * 4 + h) * Nn + n] - mx[h]) * inv[h];
    }
    __syncthreads();
    for (int nl = g; nl < 256; nl += 4) {
      float p = (float)pf[((size_t)b * Nn + nb + nl) * 64 + d];
      for (int h = 0; h < 4; ++h) acc[h] += sw[nl * 4 + h] * p;
    }
    __syncthreads();
  }
  for (int h = 0; h < 4; ++h) atomicAdd(&wsum[(b * 4 + h) * 64 + d], acc[h]);
}

// ---------------------------------------------------------------------------
// K3c: ctx = wsum @ wv_slice.T + bv;  ao = ctx @ wo.T + bo    (tiny)
// ---------------------------------------------------------------------------
__global__ __launch_bounds__(64) void k3c_ao(
    const float* __restrict__ wsum, const float* __restrict__ wv,
    const float* __restrict__ bv,   const float* __restrict__ wo,
    const float* __restrict__ bo,   float* __restrict__ ao)
{
  __shared__ float ctx[64];
  const int b = blockIdx.x, t = threadIdx.x;
  const int h = t >> 4, jj = t & 15;
  float s = bv[h * 16 + jj];
  for (int e = 0; e < 64; ++e) s += wsum[(b * 4 + h) * 64 + e] * wv[(h * 16 + jj) * 64 + e];
  ctx[t] = s;
  __syncthreads();
  float a = bo[t];
  for (int e = 0; e < 64; ++e) a += ctx[e] * wo[t * 64 + e];
  ao[b * 64 + t] = a;
}

// ---------------------------------------------------------------------------
// K4: y = pf(bf16) + ao; LayerNorm -> bf16 LDS; WMMA MLP (K=64); GELU; dot w2;
// masked scatter to output.  grid (N/256, B), block 256.
// ---------------------------------------------------------------------------
__global__ __launch_bounds__(256) void k4_refine(
    const __bf16* __restrict__ pf, const float* __restrict__ ao,
    const float* __restrict__ ln_g, const float* __restrict__ ln_b,
    const __bf16* __restrict__ w1b, const float* __restrict__ b1,
    const float* __restrict__ w2, const float* __restrict__ b2,
    const int* __restrict__ cmask, float* __restrict__ out)
{
  __shared__ __align__(16) __bf16 syn[256 * 72];
  __shared__ float s_ao[64];
  const int t = threadIdx.x, b = blockIdx.y, n0 = blockIdx.x * 256;
  if (t < 64) s_ao[t] = ao[b * 64 + t];
  __syncthreads();
  {
    const int n = n0 + t;
    const v8bf* pr = (const v8bf*)(pf + ((size_t)b * Nn + n) * 64);
    float y[64];
    float sum = 0.f, sq = 0.f;
#pragma unroll
    for (int i = 0; i < 8; ++i) {
      v8bf c = pr[i];
#pragma unroll
      for (int k = 0; k < 8; ++k) {
        float v = (float)c[k] + s_ao[i * 8 + k];
        y[i * 8 + k] = v;
        sum += v;
        sq += v * v;
      }
    }
    const float mu = sum * (1.f / 64.f);
    const float var = sq * (1.f / 64.f) - mu * mu;
    const float rstd = rsqrtf(var + 1e-5f);
#pragma unroll
    for (int e = 0; e < 64; e += 4) {
      float y0 = (y[e + 0] - mu) * rstd * ln_g[e + 0] + ln_b[e + 0];
      float y1 = (y[e + 1] - mu) * rstd * ln_g[e + 1] + ln_b[e + 1];
      float y2 = (y[e + 2] - mu) * rstd * ln_g[e + 2] + ln_b[e + 2];
      float y3 = (y[e + 3] - mu) * rstd * ln_g[e + 3] + ln_b[e + 3];
      uint2 p;
      p.x = pk_bf16(y0, y1);
      p.y = pk_bf16(y2, y3);
      *(uint2*)(&syn[t * 72 + e]) = p;
    }
  }
  __syncthreads();

  const int w = t >> 5, lane = t & 31, lr = lane & 15, hh = lane >> 4;
  float b1v[4], w2v[4];
  for (int dt = 0; dt < 4; ++dt) { b1v[dt] = b1[dt * 16 + lr]; w2v[dt] = w2[dt * 16 + lr]; }
  const float b2v = b2[0];

  for (int mt2 = 0; mt2 < 2; ++mt2) {
    const int mt = w * 2 + mt2;
    const int row = mt * 16 + lr;
    v8f acc[4] = {};
#pragma unroll
    for (int kk = 0; kk < 2; ++kk) {
      int abase = row * 72 + kk * 32 + hh * 16;
      union { struct { v8bf lo, hi; } p; v16bf v; } ua;
      ua.p.lo = *(const v8bf*)(&syn[abase]);
      ua.p.hi = *(const v8bf*)(&syn[abase + 8]);
#pragma unroll
      for (int dt = 0; dt < 4; ++dt) {
        const v16bf bf = *(const v16bf*)(&w1b[(dt * 16 + lr) * 64 + kk * 32 + hh * 16]);
        acc[dt] = __builtin_amdgcn_wmma_f32_16x16x32_bf16(
            false, ua.v, false, bf, (short)0, acc[dt], false, false);
      }
    }
#pragma unroll
    for (int r = 0; r < 8; ++r) {
      float s = 0.f;
#pragma unroll
      for (int dt = 0; dt < 4; ++dt) {
        float x = acc[dt][r] + b1v[dt];
        float gel = 0.5f * x * (1.f + erff(x * 0.70710678118f));  // exact GELU
        s += gel * w2v[dt];
      }
      s += __shfl_xor(s, 1, 32);
      s += __shfl_xor(s, 2, 32);
      s += __shfl_xor(s, 4, 32);
      s += __shfl_xor(s, 8, 32);
      if (lr == 0) {
        int n = n0 + mt * 16 + r + hh * 8;
        out[(size_t)b * Nn + n] = cmask[n] ? (s + b2v) : 0.f;
      }
    }
  }
}

// ---------------------------------------------------------------------------
extern "C" void kernel_launch(void* const* d_in, const int* in_sizes, int n_in,
                              void* d_out, int out_size, void* d_ws, size_t ws_size,
                              hipStream_t stream)
{
  (void)in_sizes; (void)n_in; (void)out_size; (void)ws_size;
  const float* X      = (const float*)d_in[0];
  const float* idemb  = (const float*)d_in[1];
  const int*   cmask  = (const int*)d_in[2];
  const float* conv_w = (const float*)d_in[3];
  const float* conv_b = (const float*)d_in[4];
  const float* idp_w  = (const float*)d_in[5];
  const float* idp_b  = (const float*)d_in[6];
  const float* wq     = (const float*)d_in[7];
  const float* bq     = (const float*)d_in[8];
  const float* wk     = (const float*)d_in[9];
  // d_in[10] = bk: drops out of softmax (row-constant shift)
  const float* wv     = (const float*)d_in[11];
  const float* bv     = (const float*)d_in[12];
  const float* wo     = (const float*)d_in[13];
  const float* bo     = (const float*)d_in[14];
  const float* ln_g   = (const float*)d_in[15];
  const float* ln_b   = (const float*)d_in[16];
  const float* w1     = (const float*)d_in[17];
  const float* b1     = (const float*)d_in[18];
  const float* w2     = (const float*)d_in[19];
  const float* b2     = (const float*)d_in[20];
  float* out = (float*)d_out;

  char* ws = (char*)d_ws;
  size_t off = 0;
  auto walloc = [&](size_t bytes) -> void* {
    void* p = ws + off;
    off += (bytes + 255) & ~(size_t)255;
    return p;
  };
  __bf16* pf       = (__bf16*)walloc((size_t)Bn * Nn * HDn * 2);  // 67 MB
  float*  scores   = (float*)walloc((size_t)Bn * NHn * Nn * 4);   // 8 MB
  float*  kq       = (float*)walloc(Bn * NHn * HDn * 4);
  __bf16* convw_bf = (__bf16*)walloc(HDn * Cc * 2);
  __bf16* w1_bf    = (__bf16*)walloc(HDn * HDn * 2);
  float*  stats    = (float*)walloc(Bn * NHn * 2 * 4);
  float*  wsum     = (float*)walloc(Bn * NHn * HDn * 4);
  float*  ao       = (float*)walloc(Bn * HDn * 4);

  k1_prep<<<1, 256, 0, stream>>>(idemb, idp_w, idp_b, wq, bq, wk, conv_w, w1,
                                 kq, convw_bf, w1_bf);
  k2_pf_scores<<<dim3(Nn / 128, Bn), 256, 0, stream>>>(X, convw_bf, conv_b, kq,
                                                       cmask, pf, scores);
  k3a_stats<<<Bn * NHn, 256, 0, stream>>>(scores, stats, wsum);
  k3b_wsum<<<dim3(8, Bn), 256, 0, stream>>>(scores, pf, stats, wsum);
  k3c_ao<<<Bn, 64, 0, stream>>>(wsum, wv, bv, wo, bo, ao);
  k4_refine<<<dim3(Nn / 256, Bn), 256, 0, stream>>>(pf, ao, ln_g, ln_b, w1_bf,
                                                    b1, w2, b2, cmask, out);
}
